// CSAB2_1322849927416
// MI455X (gfx1250) — matrix operand order repaired and checked by reading, not compiled, not run on hardware
//
#include <hip/hip_runtime.h>

typedef __attribute__((ext_vector_type(16))) __bf16 v16bf;
typedef __attribute__((ext_vector_type(8)))  float  v8f;
typedef int v4i_g __attribute__((vector_size(16)));   // matches builtin param type

#define DEV_INLINE __device__ __forceinline__
#define AS1 __attribute__((address_space(1)))
#define AS3 __attribute__((address_space(3)))

// CDNA5 async global->LDS copy (bypasses VGPRs, tracked by ASYNCcnt).
// Guarded: if the toolchain doesn't expose the builtin, fall back to the
// known-good load+ds_store staging.
#if defined(__HIP_DEVICE_COMPILE__) && \
    __has_builtin(__builtin_amdgcn_global_load_async_to_lds_b128)
#define CSAB_ASYNC 1
#else
#define CSAB_ASYNC 0
#endif

DEV_INLINE v8f wmma_bf16(v16bf a, v16bf b, v8f c) {
  // D = A(16x32 bf16) * B(32x16 bf16) + C(16x16 f32)
  return __builtin_amdgcn_wmma_f32_16x16x32_bf16(false, a, false, b, (short)0, c,
                                                 false, false);
}

DEV_INLINE void csab_stage16(const __bf16* src, __bf16* lds) {
#if CSAB_ASYNC
  __builtin_amdgcn_global_load_async_to_lds_b128(
      (AS1 v4i_g*)(void*)src, (AS3 v4i_g*)(void*)lds, 0, 0);
#else
#pragma unroll
  for (int j = 0; j < 8; ++j) lds[j] = src[j];
#endif
}

DEV_INLINE void csab_wait_async() {
#if CSAB_ASYNC
  asm volatile("s_wait_asynccnt 0" ::: "memory");
#endif
}

// ---------------------------------------------------------------------------
// fp32 -> bf16 conversion (n must be a multiple of 4; all our sizes are)
// ---------------------------------------------------------------------------
__global__ __launch_bounds__(256) void csab_cvt_kernel(const float* __restrict__ in,
                                                       __bf16* __restrict__ out, int n) {
  int i = (blockIdx.x * 256 + threadIdx.x) * 4;
  if (i + 3 < n) {
    float4 v = *(const float4*)(in + i);
    out[i + 0] = (__bf16)v.x;
    out[i + 1] = (__bf16)v.y;
    out[i + 2] = (__bf16)v.z;
    out[i + 3] = (__bf16)v.w;
  }
}

// ---------------------------------------------------------------------------
// Tiled WMMA GEMM:  C[M,N] = A[M,K] @ B[K,N] + bias
// MODE 0: out bf16;  MODE 1: out f32 with ReLU
// block 256 threads (8 waves); 128x128 tile; K-step 32
// ---------------------------------------------------------------------------
template <int MODE>
__global__ __launch_bounds__(256) void csab_gemm_kernel(
    const __bf16* __restrict__ A, const __bf16* __restrict__ Bm,
    const float* __restrict__ bias, void* __restrict__ Cout,
    int M, int N, int K) {
  const int tid = threadIdx.x;
  const int wv = tid >> 5, lane = tid & 31;
  const int lg = lane & 15, grp = lane >> 4;
  const int row0 = blockIdx.y * 128;
  const int col0 = blockIdx.x * 128;

  __shared__ __bf16 As[128][40];   // [m][k], padded
  __shared__ __bf16 Bs[128][40];   // [n][k] (transposed), padded

  v8f acc[8];
#pragma unroll
  for (int j = 0; j < 8; ++j)
#pragma unroll
    for (int e = 0; e < 8; ++e) acc[j][e] = 0.0f;

  const int m0 = wv * 16;

  for (int k0 = 0; k0 < K; k0 += 32) {
#pragma unroll
    for (int i = 0; i < 2; ++i) {
      int c = tid + i * 256;
      {  // A tile: 128 rows x 32 k, async 16B chunks straight into LDS
        int m = c >> 2, k8 = (c & 3) * 8;
        csab_stage16(A + (size_t)(row0 + m) * K + k0 + k8, &As[m][k8]);
      }
      {  // B tile: 32 k x 128 n, store transposed (element scatter)
        int kk = c >> 4, n8 = (c & 15) * 8;
        const __bf16* src = Bm + (size_t)(k0 + kk) * N + col0 + n8;
#pragma unroll
        for (int j = 0; j < 8; ++j) Bs[n8 + j][kk] = src[j];
      }
    }
    csab_wait_async();
    __syncthreads();

    // A fragment: lanes 0-15: M=lg, K 0-7 & 16-23; lanes 16-31: K 8-15 & 24-31
    v16bf af;
    const int ak = grp * 8;
#pragma unroll
    for (int i = 0; i < 8; ++i) {
      af[i]     = As[m0 + lg][ak + i];
      af[8 + i] = As[m0 + lg][16 + ak + i];
    }
#pragma unroll
    for (int j = 0; j < 8; ++j) {
      v16bf bfr;  // B fragment: N=lg (+16 tile offset), K same order as A fragment
      const int bn = j * 16 + lg;
      const int bk = grp * 8;
#pragma unroll
      for (int i = 0; i < 16; ++i) bfr[i] = Bs[bn][bk + (i & 7) + ((i >> 3) << 4)];
      acc[j] = wmma_bf16(af, bfr, acc[j]);
    }
    __syncthreads();
  }

#pragma unroll
  for (int j = 0; j < 8; ++j) {
    const int col = col0 + j * 16 + lg;
    const float bv = bias[col];
#pragma unroll
    for (int r = 0; r < 8; ++r) {
      const int row = row0 + m0 + r + grp * 8;  // D layout: M = r + 8*grp
      float v = acc[j][r] + bv;
      if (MODE == 0) {
        ((__bf16*)Cout)[(size_t)row * N + col] = (__bf16)v;
      } else {
        ((float*)Cout)[(size_t)row * N + col] = fmaxf(v, 0.0f);
      }
    }
  }
}

// ---------------------------------------------------------------------------
// Fused flash-style attention for one (batch, head, variant, 128-q-row block).
// Q,K,V are bf16 (B,N,1024) with head h at cols [h*128, h*128+128).
// O (fp32, (B,N,1024)) = Qh + softmax(Qh Kh^T / 32) Vh
// ---------------------------------------------------------------------------
__global__ __launch_bounds__(256) void csab_attn_kernel(
    const __bf16* __restrict__ Qx, const __bf16* __restrict__ Kx, const __bf16* __restrict__ Vx,
    const __bf16* __restrict__ Qy, const __bf16* __restrict__ Ky, const __bf16* __restrict__ Vy,
    float* __restrict__ O0, float* __restrict__ O1,
    float* __restrict__ O2, float* __restrict__ O3) {
  const int tid = threadIdx.x;
  const int wv = tid >> 5, lane = tid & 31;
  const int lg = lane & 15, grp = lane >> 4;
  const int z = blockIdx.z;
  const int bh = blockIdx.y;
  const int b = bh >> 3, h = bh & 7;
  const int q0 = blockIdx.x * 128;

  const __bf16* Qp = (z < 2) ? Qx : Qy;
  const __bf16* Kp = (z & 1) ? Ky : Kx;
  const __bf16* Vp = (z & 1) ? Vy : Vx;
  float* Op = (z == 0) ? O0 : (z == 1) ? O1 : (z == 2) ? O2 : O3;

  __shared__ __bf16 Qs[128][136];      // [q][d]
  __shared__ __bf16 Ks[128][136];      // [key][d]   (natural = B-layout for S)
  __shared__ __bf16 Vs[128][136];      // [d][key]   (transposed = B-layout for O)
  __shared__ __bf16 Ps[8][16][136];    // per-wave P round-trip (C -> A layout)

  const size_t headOff = (size_t)h * 128;
  const size_t qbase = ((size_t)b * 1024 + q0) * 1024 + headOff;

  // stage Q tile (128 x 128) via async copy
#pragma unroll
  for (int i = 0; i < 8; ++i) {
    int c = tid + i * 256;
    int m = c >> 4, d8 = (c & 15) * 8;
    csab_stage16(Qp + qbase + (size_t)m * 1024 + d8, &Qs[m][d8]);
  }
  csab_wait_async();
  __syncthreads();

  // preload 4 Q A-fragments (K-dim = 128 = 4 x 32) for this wave's 16 rows
  const int m0 = wv * 16;
  v16bf qa[4];
#pragma unroll
  for (int dk = 0; dk < 4; ++dk) {
    const int kb8 = dk * 32 + grp * 8;
#pragma unroll
    for (int i = 0; i < 8; ++i) {
      qa[dk][i]     = Qs[m0 + lg][kb8 + i];
      qa[dk][8 + i] = Qs[m0 + lg][kb8 + 16 + i];
    }
  }

  v8f o[8];
  float rm[8], rl[8];
#pragma unroll
  for (int j = 0; j < 8; ++j) {
    rm[j] = -3.0e38f;
    rl[j] = 0.0f;
#pragma unroll
    for (int e = 0; e < 8; ++e) o[j][e] = 0.0f;
  }

  for (int kb = 0; kb < 8; ++kb) {
    const size_t kvbase = ((size_t)b * 1024 + kb * 128) * 1024 + headOff;
#pragma unroll
    for (int i = 0; i < 8; ++i) {
      int c = tid + i * 256;
      int kk = c >> 4, d8 = (c & 15) * 8;
      // K: natural layout -> async straight into LDS
      csab_stage16(Kp + kvbase + (size_t)kk * 1024 + d8, &Ks[kk][d8]);
      // V: transposed layout -> manual scatter
      const __bf16* vsrc = Vp + kvbase + (size_t)kk * 1024 + d8;
#pragma unroll
      for (int j = 0; j < 8; ++j) Vs[d8 + j][kk] = vsrc[j];
    }
    csab_wait_async();
    __syncthreads();

    // prefetch next K/V block into cache while we compute on this one
    if (kb + 1 < 8) {
      const size_t nb = ((size_t)b * 1024 + (kb + 1) * 128) * 1024 + headOff;
      int kk = tid >> 4, d8 = (tid & 15) * 8;
      __builtin_prefetch(Kp + nb + (size_t)kk * 1024 + d8, 0, 3);
      __builtin_prefetch(Vp + nb + (size_t)kk * 1024 + d8, 0, 3);
    }

    // S (16 x 128) = Q Kh^T, 8 key tiles x 4 d-chunks
    v8f st[8];
#pragma unroll
    for (int j = 0; j < 8; ++j) {
      v8f s;
#pragma unroll
      for (int e = 0; e < 8; ++e) s[e] = 0.0f;
#pragma unroll
      for (int dk = 0; dk < 4; ++dk) {
        v16bf kf;
        const int krow = j * 16 + lg;
        const int kk8 = dk * 32 + grp * 8;
#pragma unroll
        for (int i = 0; i < 16; ++i) kf[i] = Ks[krow][kk8 + (i & 7) + ((i >> 3) << 4)];
        s = wmma_bf16(qa[dk], kf, s);
      }
#pragma unroll
      for (int e = 0; e < 8; ++e) s[e] *= 0.03125f;  // 1/sqrt(1024)
      st[j] = s;
    }

    // online softmax update; row r of acc = local row (r + 8*grp)
#pragma unroll
    for (int r = 0; r < 8; ++r) {
      float mx = -3.0e38f;
#pragma unroll
      for (int j = 0; j < 8; ++j) mx = fmaxf(mx, st[j][r]);
#pragma unroll
      for (int off = 1; off < 16; off <<= 1) mx = fmaxf(mx, __shfl_xor(mx, off, 32));
      const float mnew = fmaxf(rm[r], mx);
      const float alpha = __expf(rm[r] - mnew);
      rm[r] = mnew;
      float ps = 0.0f;
#pragma unroll
      for (int j = 0; j < 8; ++j) {
        float p = __expf(st[j][r] - mnew);
        st[j][r] = p;
        ps += p;
      }
#pragma unroll
      for (int off = 1; off < 16; off <<= 1) ps += __shfl_xor(ps, off, 32);
      rl[r] = rl[r] * alpha + ps;
#pragma unroll
      for (int j = 0; j < 8; ++j) o[j][r] *= alpha;
    }

    // P (C-layout regs) -> per-wave LDS -> A-layout fragments
#pragma unroll
    for (int j = 0; j < 8; ++j)
#pragma unroll
      for (int r = 0; r < 8; ++r)
        Ps[wv][r + grp * 8][j * 16 + lg] = (__bf16)st[j][r];

    // O (16 x 128) += P (16 x 128) @ V (128 x 128)
#pragma unroll
    for (int dk = 0; dk < 4; ++dk) {
      v16bf pa;
      const int kb8 = dk * 32 + grp * 8;
#pragma unroll
      for (int i = 0; i < 8; ++i) {
        pa[i]     = Ps[wv][lg][kb8 + i];
        pa[8 + i] = Ps[wv][lg][kb8 + 16 + i];
      }
#pragma unroll
      for (int j = 0; j < 8; ++j) {
        v16bf vf;
        const int drow = j * 16 + lg;
        const int kk8 = dk * 32 + grp * 8;
#pragma unroll
        for (int i = 0; i < 16; ++i) vf[i] = Vs[drow][kk8 + (i & 7) + ((i >> 3) << 4)];
        o[j] = wmma_bf16(pa, vf, o[j]);
      }
    }
    __syncthreads();
  }

  // normalize, add per-head Q residual, store fp32
#pragma unroll
  for (int j = 0; j < 8; ++j) {
#pragma unroll
    for (int r = 0; r < 8; ++r) {
      const int m = m0 + r + grp * 8;
      const int d = j * 16 + lg;
      float val = o[j][r] / rl[r] + (float)Qs[m][d];
      Op[((size_t)b * 1024 + q0 + m) * 1024 + headOff + d] = val;
    }
  }
}

// ---------------------------------------------------------------------------
// LayerNorm over last dim (1024). Variant A: bf16 out into concat buffer
// (row stride 2048, column offset). Variant B: f32 out (row stride 1024).
// ---------------------------------------------------------------------------
DEV_INLINE void ln_stats(const float* xr, int tid, float x[4], float& mu, float& rs) {
  float s = 0.0f, sq = 0.0f;
#pragma unroll
  for (int i = 0; i < 4; ++i) {
    x[i] = xr[tid + i * 256];
    s += x[i];
    sq += x[i] * x[i];
  }
#pragma unroll
  for (int off = 1; off < 32; off <<= 1) {
    s += __shfl_xor(s, off, 32);
    sq += __shfl_xor(sq, off, 32);
  }
  __shared__ float ss[8], ssq[8];
  __shared__ float mu_s, rs_s;
  if ((tid & 31) == 0) { ss[tid >> 5] = s; ssq[tid >> 5] = sq; }
  __syncthreads();
  if (tid == 0) {
    float S = 0.0f, Q2 = 0.0f;
#pragma unroll
    for (int w = 0; w < 8; ++w) { S += ss[w]; Q2 += ssq[w]; }
    float m = S * (1.0f / 1024.0f);
    float var = Q2 * (1.0f / 1024.0f) - m * m;
    mu_s = m;
    rs_s = rsqrtf(var + 1e-5f);
  }
  __syncthreads();
  mu = mu_s;
  rs = rs_s;
}

__global__ __launch_bounds__(256) void csab_ln_cat_kernel(
    const float* __restrict__ X, const float* __restrict__ g,
    const float* __restrict__ be, __bf16* __restrict__ out, int colOff) {
  const int row = blockIdx.x, tid = threadIdx.x;
  float x[4], mu, rs;
  ln_stats(X + (size_t)row * 1024, tid, x, mu, rs);
#pragma unroll
  for (int i = 0; i < 4; ++i) {
    int c = tid + i * 256;
    float y = (x[i] - mu) * rs * g[c] + be[c];
    out[(size_t)row * 2048 + colOff + c] = (__bf16)y;
  }
}

__global__ __launch_bounds__(256) void csab_ln_out_kernel(
    const float* __restrict__ X, const float* __restrict__ g,
    const float* __restrict__ be, float* __restrict__ out) {
  const int row = blockIdx.x, tid = threadIdx.x;
  float x[4], mu, rs;
  ln_stats(X + (size_t)row * 1024, tid, x, mu, rs);
#pragma unroll
  for (int i = 0; i < 4; ++i) {
    int c = tid + i * 256;
    out[(size_t)row * 1024 + c] = (x[i] - mu) * rs * g[c] + be[c];
  }
}

// ---------------------------------------------------------------------------
extern "C" void kernel_launch(void* const* d_in, const int* in_sizes, int n_in,
                              void* d_out, int out_size, void* d_ws, size_t ws_size,
                              hipStream_t stream) {
  (void)in_sizes; (void)n_in; (void)out_size; (void)ws_size;
  const float* X   = (const float*)d_in[0];
  const float* Y   = (const float*)d_in[1];
  const float* Wq_[3] = {(const float*)d_in[2], (const float*)d_in[4], (const float*)d_in[6]};
  const float* bX_[3] = {(const float*)d_in[3], (const float*)d_in[5], (const float*)d_in[7]};
  const float* Wy_[3] = {(const float*)d_in[8], (const float*)d_in[10], (const float*)d_in[12]};
  const float* bY_[3] = {(const float*)d_in[9], (const float*)d_in[11], (const float*)d_in[13]};
  const float* WXw = (const float*)d_in[14]; const float* bXv = (const float*)d_in[15];
  const float* WYw = (const float*)d_in[16]; const float* bYv = (const float*)d_in[17];
  const float* g0 = (const float*)d_in[18]; const float* b0 = (const float*)d_in[19];
  const float* g1 = (const float*)d_in[20]; const float* b1 = (const float*)d_in[21];

  char* ws = (char*)d_ws;
  size_t cur = 0;
  auto take = [&](size_t bytes) -> char* {
    cur = (cur + 255) & ~(size_t)255;
    char* p = ws + cur;
    cur += bytes;
    return p;
  };
  const size_t NE = 4096ull * 1024;  // tokens * DV

  __bf16* Xbf = (__bf16*)take(NE * 2);
  __bf16* Ybf = (__bf16*)take(NE * 2);
  __bf16* WbfX[3]; __bf16* WbfY[3];
  for (int i = 0; i < 3; ++i) WbfX[i] = (__bf16*)take(1024ull * 1024 * 2);
  for (int i = 0; i < 3; ++i) WbfY[i] = (__bf16*)take(1024ull * 1024 * 2);
  __bf16* WXbf = (__bf16*)take(2048ull * 1024 * 2);
  __bf16* WYbf = (__bf16*)take(2048ull * 1024 * 2);
  __bf16* Qx = (__bf16*)take(NE * 2); __bf16* Kx = (__bf16*)take(NE * 2);
  __bf16* Vx = (__bf16*)take(NE * 2);
  __bf16* Qy = (__bf16*)take(NE * 2); __bf16* Ky = (__bf16*)take(NE * 2);
  __bf16* Vy = (__bf16*)take(NE * 2);
  float* O0 = (float*)take(NE * 4); float* O1 = (float*)take(NE * 4);
  float* O2 = (float*)take(NE * 4); float* O3 = (float*)take(NE * 4);
  __bf16* CatX = (__bf16*)take(4096ull * 2048 * 2);
  __bf16* CatY = (__bf16*)take(4096ull * 2048 * 2);
  float* Hx = (float*)take(NE * 4);
  float* Hy = (float*)take(NE * 4);

  // 1) cast inputs + weights to bf16
  csab_cvt_kernel<<<(int)(NE / 1024), 256, 0, stream>>>(X, Xbf, (int)NE);
  csab_cvt_kernel<<<(int)(NE / 1024), 256, 0, stream>>>(Y, Ybf, (int)NE);
  for (int i = 0; i < 3; ++i) {
    csab_cvt_kernel<<<1024, 256, 0, stream>>>(Wq_[i], WbfX[i], 1024 * 1024);
    csab_cvt_kernel<<<1024, 256, 0, stream>>>(Wy_[i], WbfY[i], 1024 * 1024);
  }
  csab_cvt_kernel<<<2048, 256, 0, stream>>>(WXw, WXbf, 2048 * 1024);
  csab_cvt_kernel<<<2048, 256, 0, stream>>>(WYw, WYbf, 2048 * 1024);

  // 2) six projection GEMMs (4096 x 1024 x 1024), bf16 out
  dim3 gg(8, 32);
  csab_gemm_kernel<0><<<gg, 256, 0, stream>>>(Xbf, WbfX[0], bX_[0], Qx, 4096, 1024, 1024);
  csab_gemm_kernel<0><<<gg, 256, 0, stream>>>(Xbf, WbfX[1], bX_[1], Kx, 4096, 1024, 1024);
  csab_gemm_kernel<0><<<gg, 256, 0, stream>>>(Xbf, WbfX[2], bX_[2], Vx, 4096, 1024, 1024);
  csab_gemm_kernel<0><<<gg, 256, 0, stream>>>(Ybf, WbfY[0], bY_[0], Qy, 4096, 1024, 1024);
  csab_gemm_kernel<0><<<gg, 256, 0, stream>>>(Ybf, WbfY[1], bY_[1], Ky, 4096, 1024, 1024);
  csab_gemm_kernel<0><<<gg, 256, 0, stream>>>(Ybf, WbfY[2], bY_[2], Vy, 4096, 1024, 1024);

  // 3) four fused attentions (z: xx, xy, yx, yy)
  dim3 ga(8, 32, 4);
  csab_attn_kernel<<<ga, 256, 0, stream>>>(Qx, Kx, Vx, Qy, Ky, Vy, O0, O1, O2, O3);

  // 4) LayerNorm(g0,b0) -> bf16 concat buffers
  csab_ln_cat_kernel<<<4096, 256, 0, stream>>>(O0, g0, b0, CatX, 0);
  csab_ln_cat_kernel<<<4096, 256, 0, stream>>>(O1, g0, b0, CatX, 1024);
  csab_ln_cat_kernel<<<4096, 256, 0, stream>>>(O2, g0, b0, CatY, 0);
  csab_ln_cat_kernel<<<4096, 256, 0, stream>>>(O3, g0, b0, CatY, 1024);

  // 5) output projections (4096 x 1024 x 2048) + ReLU, f32 out
  csab_gemm_kernel<1><<<gg, 256, 0, stream>>>(CatX, WXbf, bXv, Hx, 4096, 1024, 2048);
  csab_gemm_kernel<1><<<gg, 256, 0, stream>>>(CatY, WYbf, bYv, Hy, 4096, 1024, 2048);

  // 6) final LayerNorm(g1,b1) -> d_out (O_x then O_y, fp32)
  float* outX = (float*)d_out;
  float* outY = outX + NE;
  csab_ln_out_kernel<<<4096, 256, 0, stream>>>(Hx, g1, b1, outX);
  csab_ln_out_kernel<<<4096, 256, 0, stream>>>(Hy, g1, b1, outY);
}
